// GeneratorNetworkDiscrete_6468220748212
// MI455X (gfx1250) — matrix operand order repaired and checked by reading, not compiled, not run on hardware
//
#include <hip/hip_runtime.h>
#include <hip/hip_bf16.h>
#include <stdint.h>

// ---------------------------------------------------------------------------
// CDNA5 / gfx1250 persistent recurrent-GEMM kernel.
// wave32, WMMA 16x16x32 bf16, recurrent state resident in LDS (320KB/WGP).
// ---------------------------------------------------------------------------

typedef __bf16 bf16;
typedef __attribute__((ext_vector_type(16))) __bf16 v16bf;
typedef __attribute__((ext_vector_type(8)))  __bf16 v8bf;
typedef __attribute__((ext_vector_type(8)))  float  v8f;

#define BT   32      // batch rows per workgroup
#define NW   8       // waves per workgroup (256 threads, wave32)
#define LSX  520     // bf16 row stride for X/H LDS regions (1040B: 16B aligned, bank-friendly)
#define LSU  168     // bf16 row stride for U (latent||prev||pad, K=160)
#define RNNU 512
#define GDIM 2048    // 4*RNN

// ---- LDS byte offsets (total 316,416 B < 320KB) ----
#define OFF_X   0
#define OFF_H0  (OFF_X  + BT*LSX*2)            // double buffered
#define OFF_H1  (OFF_H0 + 2*BT*LSX*2)
#define OFF_C0  (OFF_H1 + 2*BT*LSX*2)
#define OFF_C1  (OFF_C0 + BT*RNNU*4)
#define OFF_U   (OFF_C1 + BT*RNNU*4)
#define OFF_LOG (OFF_U  + BT*LSU*2)
#define SMEM_BYTES (OFF_LOG + BT*64*4)

// ---- fast branch-free activations (native trans ops; no EXEC divergence) ----
__device__ __forceinline__ float rcp_fast(float x) { return __builtin_amdgcn_rcpf(x); }

__device__ __forceinline__ float sig_fast(float x) {
  return rcp_fast(1.0f + __expf(-x));          // v_exp_f32 + v_rcp_f32
}

__device__ __forceinline__ float tanh_fast(float x) {
#if defined(__gfx1250__) && __has_builtin(__builtin_amdgcn_tanhf)
  return __builtin_amdgcn_tanhf(x);            // native V_TANH_F32
#else
  float ax = __builtin_fabsf(x);
  float e  = __expf(ax * -2.0f);               // underflows to 0 for large |x| -> r=1
  float r  = (1.0f - e) * rcp_fast(1.0f + e);
  return __builtin_copysignf(r, x);
#endif
}

__device__ __forceinline__ uint32_t rotl32(uint32_t v, int n) { return (v << n) | (v >> (32 - n)); }

// JAX threefry2x32 (key = PRNGKey(42) -> (0, 42)), uniform(1e-8, 1.0), gumbel.
__device__ __forceinline__ float gumbel_at(uint64_t idx, uint64_t half) {
  bool second = idx >= half;
  uint32_t p  = (uint32_t)(second ? idx - half : idx);
  uint32_t x0 = p, x1 = p + (uint32_t)half;
  const uint32_t ks0 = 0u, ks1 = 42u, ks2 = ks0 ^ ks1 ^ 0x1BD11BDAu;
  x0 += ks0; x1 += ks1;
#define TFR(r) { x0 += x1; x1 = rotl32(x1, r); x1 ^= x0; }
  TFR(13) TFR(15) TFR(26) TFR(6)   x0 += ks1; x1 += ks2 + 1u;
  TFR(17) TFR(29) TFR(16) TFR(24)  x0 += ks2; x1 += ks0 + 2u;
  TFR(13) TFR(15) TFR(26) TFR(6)   x0 += ks0; x1 += ks1 + 3u;
  TFR(17) TFR(29) TFR(16) TFR(24)  x0 += ks1; x1 += ks2 + 4u;
  TFR(13) TFR(15) TFR(26) TFR(6)   x0 += ks2; x1 += ks0 + 5u;
#undef TFR
  uint32_t bits = second ? x1 : x0;
  float u = __uint_as_float((bits >> 9) | 0x3F800000u) - 1.0f;  // [0,1)
  u = u * (1.0f - 1e-8f) + 1e-8f;
  u = fmaxf(u, 1e-8f);
  return -__logf(-__logf(u));
}

// A fragment (16x32 bf16, MxK) from LDS, per documented CDNA5 layout:
// lanes 0-15: row=lane, VGPR0-3 K=k0..k0+7, VGPR4-7 K=k0+16..k0+23 (k0 = kc*32)
// lanes 16-31: same rows, K base +8.
__device__ __forceinline__ v16bf load_a(const bf16* __restrict__ base, int ldr, int m0, int kc) {
  int lane = threadIdx.x & 31;
  const bf16* p = base + (size_t)(m0 + (lane & 15)) * ldr + kc * 32 + ((lane >> 4) << 3);
  v8bf lo = *(const v8bf*)(p);
  v8bf hi = *(const v8bf*)(p + 16);
  v16bf a;
#pragma unroll
  for (int i = 0; i < 8; ++i) { a[i] = lo[i]; a[i + 8] = hi[i]; }
  return a;
}

// B fragment: weights pre-packed in exact fragment order -> one 32B load/lane.
__device__ __forceinline__ v16bf load_b(const bf16* __restrict__ wpk, int kc, int nt, int ntiles) {
  int lane = threadIdx.x & 31;
  const bf16* p = wpk + ((size_t)(kc * ntiles + nt) << 9) + (lane << 4);
  return *(const v16bf*)(p);
}

__device__ __forceinline__ v8f wmma_bf16(v16bf a, v16bf b, v8f c) {
  return __builtin_amdgcn_wmma_f32_16x16x32_bf16(false, a, false, b, (short)0, c,
                                                 false, false);
}

// One LSTM cell for the whole BT-row tile. A = [A1 | A2] (K = 1024).
// Each macro-tile = 16 rows x 16 units; its 4 gate tiles (i,f,g,o) share A frags.
__device__ __forceinline__ void lstm_cell(const bf16* __restrict__ A1,
                                          const bf16* __restrict__ A2,
                                          bf16* __restrict__ Hout,
                                          float* __restrict__ C,
                                          const bf16* __restrict__ wpk,
                                          const float* __restrict__ bias,
                                          int wave, int lane) {
  for (int t = wave; t < (BT / 16) * (RNNU / 16); t += NW) {
    int rt = t >> 5, ut = t & 31;
    v8f acc[4] = {v8f{}, v8f{}, v8f{}, v8f{}};
    for (int kc = 0; kc < 32; ++kc) {
      v16bf a = (kc < 16) ? load_a(A1, LSX, rt * 16, kc)
                          : load_a(A2, LSX, rt * 16, kc - 16);
      if (kc + 1 < 32)  // hint next K-slice of packed weights into cache
        __builtin_prefetch(wpk + (((size_t)(kc + 1) * 128 + ut) << 9), 0, 0);
#pragma unroll
      for (int g = 0; g < 4; ++g) {
        v16bf b = load_b(wpk, kc, g * 32 + ut, 128);
        acc[g] = wmma_bf16(a, b, acc[g]);
      }
    }
    int j  = ut * 16 + (lane & 15);
    int mb = rt * 16 + ((lane >> 4) << 3);
    float bi = bias[0 * RNNU + j], bff = bias[1 * RNNU + j];
    float bg = bias[2 * RNNU + j], bo  = bias[3 * RNNU + j];
#pragma unroll
    for (int v = 0; v < 8; ++v) {
      int m = mb + v;
      float ig = sig_fast(acc[0][v] + bi);
      float fg = sig_fast(acc[1][v] + bff);
      float gg = tanh_fast(acc[2][v] + bg);
      float og = sig_fast(acc[3][v] + bo);
      float c  = fg * C[(size_t)m * RNNU + j] + ig * gg;
      C[(size_t)m * RNNU + j] = c;
      Hout[(size_t)m * LSX + j] = (bf16)(og * tanh_fast(c));
    }
  }
}

extern "C" __global__ __launch_bounds__(256)
void gen_rnn_main(const float* __restrict__ latent,
                  const float* __restrict__ b_unpack,
                  const float* __restrict__ b_angle,
                  const bf16* __restrict__ wup,
                  const bf16* __restrict__ wg0,
                  const bf16* __restrict__ wg1,
                  const bf16* __restrict__ wang,
                  const float* __restrict__ bias0,
                  const float* __restrict__ bias1,
                  float* __restrict__ out,
                  const int* __restrict__ nseg,
                  int B) {
  extern __shared__ char smem[];
  bf16*  X   = (bf16*)(smem + OFF_X);
  bf16*  H0  = (bf16*)(smem + OFF_H0);
  bf16*  H1  = (bf16*)(smem + OFF_H1);
  float* C0  = (float*)(smem + OFF_C0);
  float* C1  = (float*)(smem + OFF_C1);
  bf16*  U   = (bf16*)(smem + OFF_U);
  float* LOG = (float*)(smem + OFF_LOG);

  const int tid = threadIdx.x, wave = tid >> 5, lane = tid & 31;
  const int b0  = blockIdx.x * BT;
  const int S   = *nseg;
  const uint64_t half = ((uint64_t)S * (uint64_t)B * 64ull) >> 1;

  // init: U = [latent | 0 | 0pad], H/C = 0
  for (int i = tid; i < BT * LSU; i += 256) {
    int r = i / LSU, c = i % LSU;
    float v = (c < 128) ? latent[(size_t)(b0 + r) * 128 + c] : 0.0f;
    U[i] = (bf16)v;
  }
  for (int i = tid; i < 2 * BT * LSX; i += 256) { H0[i] = (bf16)0.0f; H1[i] = (bf16)0.0f; }
  for (int i = tid; i < BT * RNNU; i += 256)    { C0[i] = 0.0f; C1[i] = 0.0f; }
  __syncthreads();

  int cur = 0;
  for (int s = 0; s < S; ++s) {
    // ---- unpack: X = tanh(U @ Wup^T + b_unpack), K=160 (130 real) ----
    for (int t = wave; t < (BT / 16) * (RNNU / 16); t += NW) {
      int rt = t >> 5, nt = t & 31;
      v8f acc = v8f{};
      for (int kc = 0; kc < 5; ++kc) {
        v16bf a = load_a(U, LSU, rt * 16, kc);
        v16bf b = load_b(wup, kc, nt, 32);
        acc = wmma_bf16(a, b, acc);
      }
      int j  = nt * 16 + (lane & 15);
      int mb = rt * 16 + ((lane >> 4) << 3);
      float bb = b_unpack[j];
#pragma unroll
      for (int v = 0; v < 8; ++v)
        X[(size_t)(mb + v) * LSX + j] = (bf16)tanh_fast(acc[v] + bb);
    }
    __syncthreads();

    // ---- LSTM cell 0: A = [X | H0_cur] -> H0_nxt, C0 ----
    lstm_cell(X, H0 + cur * BT * LSX, H0 + (1 - cur) * BT * LSX, C0, wg0, bias0, wave, lane);
    __syncthreads();

    // ---- LSTM cell 1: A = [H0_nxt | H1_cur] -> H1_nxt, C1 ----
    lstm_cell(H0 + (1 - cur) * BT * LSX, H1 + cur * BT * LSX,
              H1 + (1 - cur) * BT * LSX, C1, wg1, bias1, wave, lane);
    __syncthreads();

    // ---- logits = H1_nxt @ W_angle^T + b_angle, [BT,64] ----
    {
      int rt = wave >> 2, nt = wave & 3;  // 8 tiles, one per wave
      v8f acc = v8f{};
      const bf16* A = H1 + (1 - cur) * BT * LSX;
      for (int kc = 0; kc < 16; ++kc) {
        v16bf a = load_a(A, LSX, rt * 16, kc);
        v16bf b = load_b(wang, kc, nt, 4);
        acc = wmma_bf16(a, b, acc);
      }
      int n  = nt * 16 + (lane & 15);
      int mb = rt * 16 + ((lane >> 4) << 3);
      float bb = b_angle[n];
#pragma unroll
      for (int v = 0; v < 8; ++v) LOG[(mb + v) * 64 + n] = acc[v] + bb;
    }
    __syncthreads();

    // ---- gumbel + hard argmax: angle = space[argmax(logits + g)] ----
    {
      int row   = wave * 4 + (lane >> 3);  // 8 lanes per row
      int dbase = (lane & 7) * 8;
      int gb    = b0 + row;
      float bestv = -3.4e38f; int besti = 0;
#pragma unroll
      for (int q = 0; q < 8; ++q) {
        int d = dbase + q;
        float g   = gumbel_at(((uint64_t)s * (uint64_t)B + (uint64_t)gb) * 64ull + d, half);
        float val = LOG[row * 64 + d] + g;
        if (val > bestv) { bestv = val; besti = d; }
      }
#pragma unroll
      for (int off = 1; off < 8; off <<= 1) {
        float ov = __shfl_xor(bestv, off, 32);
        int   oi = __shfl_xor(besti, off, 32);
        if (ov > bestv || (ov == bestv && oi < besti)) { bestv = ov; besti = oi; }
      }
      if ((lane & 7) == 0) {
        float ang = -1.0f + 2.0f * (float)besti / 63.0f;
        U[row * LSU + 128] = (bf16)ang;                         // prev feedback
        out[((size_t)gb * S + s) * 2 + 0] = ang;
        out[((size_t)gb * S + s) * 2 + 1] = 0.0f;
      }
    }
    cur ^= 1;
    __syncthreads();
  }
}

// ---------------------------------------------------------------------------
// Prep: pack W (gates = x @ W^T) into per-lane WMMA B-fragment order.
// B(k,n) = k<Ka ? A[n*Ka+k] : (k<Ka+Kb ? Bsrc[n*Kb+(k-Ka)] : 0).
// Packed tile (kc,nt): 32 lanes x 16 bf16; lane l -> col n=nt*16+(l&15),
// K = kc*32 + (e<8 ? 0 : 16) + (l<16 ? 0 : 8) + (e&7), e=0..15.
// ---------------------------------------------------------------------------
extern "C" __global__ void pack_wmma_b(bf16* __restrict__ dst,
                                       const float* __restrict__ A, int Ka,
                                       const float* __restrict__ Bsrc, int Kb,
                                       int N, int Ktot) {
  int ntiles = N >> 4, kcnt = Ktot >> 5;
  size_t total = (size_t)kcnt * ntiles * 512;
  for (size_t i = (size_t)blockIdx.x * blockDim.x + threadIdx.x; i < total;
       i += (size_t)gridDim.x * blockDim.x) {
    size_t tile = i >> 9;
    int r = (int)(i & 511);
    int lane = r >> 4, e = r & 15;
    int kc = (int)(tile / ntiles), nt = (int)(tile % ntiles);
    int n = nt * 16 + (lane & 15);
    int k = kc * 32 + ((e >> 3) << 4) + ((lane >> 4) << 3) + (e & 7);
    float v = 0.0f;
    if (k < Ka)               v = A[(size_t)n * Ka + k];
    else if (k < Ka + Kb)     v = Bsrc[(size_t)n * Kb + (k - Ka)];
    dst[i] = (bf16)v;
  }
}

extern "C" __global__ void combine_bias(float* __restrict__ dst,
                                        const float* __restrict__ a,
                                        const float* __restrict__ b, int n) {
  int i = blockIdx.x * blockDim.x + threadIdx.x;
  if (i < n) dst[i] = a[i] + b[i];
}

// ---------------------------------------------------------------------------
extern "C" void kernel_launch(void* const* d_in, const int* in_sizes, int n_in,
                              void* d_out, int out_size, void* d_ws, size_t ws_size,
                              hipStream_t stream) {
  const float* latent   = (const float*)d_in[0];
  const float* W_unpack = (const float*)d_in[1];
  const float* b_unpack = (const float*)d_in[2];
  const float* Wih0     = (const float*)d_in[3];
  const float* Whh0     = (const float*)d_in[4];
  const float* bih0     = (const float*)d_in[5];
  const float* bhh0     = (const float*)d_in[6];
  const float* Wih1     = (const float*)d_in[7];
  const float* Whh1     = (const float*)d_in[8];
  const float* bih1     = (const float*)d_in[9];
  const float* bhh1     = (const float*)d_in[10];
  const float* W_angle  = (const float*)d_in[11];
  const float* b_angle  = (const float*)d_in[12];
  const int*   nseg     = (const int*)d_in[13];

  const int B = in_sizes[0] / 128;  // 4096

  // workspace layout (~8.6 MB)
  const size_t WUP_E = (size_t)5 * 32 * 512;     // K=160, N=512
  const size_t WG_E  = (size_t)32 * 128 * 512;   // K=1024, N=2048
  const size_t WAN_E = (size_t)16 * 4 * 512;     // K=512,  N=64
  bf16* wup  = (bf16*)d_ws;
  bf16* wg0  = wup + WUP_E;
  bf16* wg1  = wg0 + WG_E;
  bf16* wang = wg1 + WG_E;
  float* bias0 = (float*)(wang + WAN_E);
  float* bias1 = bias0 + 2048;

  // --- pack weights into WMMA B-fragment order (bf16) ---
  pack_wmma_b<<<(int)((WUP_E + 255) / 256), 256, 0, stream>>>(
      wup, W_unpack, 130, (const float*)nullptr, 0, 512, 160);
  pack_wmma_b<<<4096, 256, 0, stream>>>(wg0, Wih0, 512, Whh0, 512, 2048, 1024);
  pack_wmma_b<<<4096, 256, 0, stream>>>(wg1, Wih1, 512, Whh1, 512, 2048, 1024);
  pack_wmma_b<<<(int)((WAN_E + 255) / 256), 256, 0, stream>>>(
      wang, W_angle, 512, (const float*)nullptr, 0, 64, 512);
  combine_bias<<<8, 256, 0, stream>>>(bias0, bih0, bhh0, 2048);
  combine_bias<<<8, 256, 0, stream>>>(bias1, bih1, bhh1, 2048);

  // --- persistent recurrent kernel: one WG per 32 batch rows, 64 steps ---
  dim3 grid(B / BT), block(256);
  gen_rnn_main<<<grid, block, SMEM_BYTES, stream>>>(
      latent, b_unpack, b_angle, wup, wg0, wg1, wang, bias0, bias1,
      (float*)d_out, nseg, B);
}
